// TransformerLayer_16295105921168
// MI455X (gfx1250) — matrix operand order repaired and checked by reading, not compiled, not run on hardware
//
#include <hip/hip_runtime.h>
#include <hip/hip_bf16.h>

// ---------------------------------------------------------------------------
// Transformer layer for MI455X (gfx1250), wave32, WMMA bf16 (f32 accumulate).
// GEMMs use double-buffered LDS staging via GLOBAL_LOAD_ASYNC_TO_LDS_B128
// (ASYNCcnt) when the toolchain exposes the builtin; otherwise a synchronous
// vector-copy fallback. Workspace requirement: ~88 MB.
// ---------------------------------------------------------------------------

#define DIMX   1024
#define SEQX   1024
#define BATCHX 4
#define NHEADS 16
#define HEADD  64
#define MLPX   4096
#define ROWSX  (BATCHX * SEQX)   // 4096

typedef __attribute__((ext_vector_type(16))) __bf16 v16bf;
typedef __attribute__((ext_vector_type(8)))  __bf16 v8bf;
typedef __attribute__((ext_vector_type(8)))  float  v8f;

// exact parameter types of the async-LDS builtin (from compiler diagnostic):
// (int4 AS1*, int4 AS3*, imm int offset, imm int cpol)
typedef int vint4 __attribute__((__vector_size__(16)));
typedef __attribute__((address_space(1))) vint4 as1_vint4;
typedef __attribute__((address_space(3))) vint4 as3_vint4;

#if defined(__gfx1250__) &&                                        \
    __has_builtin(__builtin_amdgcn_global_load_async_to_lds_b128) && \
    __has_builtin(__builtin_amdgcn_s_wait_asynccnt)
#define USE_ASYNC_LDS 1
#else
#define USE_ASYNC_LDS 0
#endif

__device__ __forceinline__ v8f v8f_zero() {
  v8f z = {0.f, 0.f, 0.f, 0.f, 0.f, 0.f, 0.f, 0.f};
  return z;
}

// 16-byte global -> LDS copy. Async (ASYNCcnt-tracked, bypasses VGPRs) when
// available; synchronous vector copy otherwise.
__device__ __forceinline__ void cp16_g2l(const __bf16* g, __bf16* l) {
#if USE_ASYNC_LDS
  __builtin_amdgcn_global_load_async_to_lds_b128(
      (as1_vint4*)g, (as3_vint4*)l, 0, 0);
#else
  *(v8bf*)l = *(const v8bf*)g;
#endif
}

__device__ __forceinline__ void cp_fence() {
#if USE_ASYNC_LDS
  __builtin_amdgcn_s_wait_asynccnt(0);   // s_wait_asynccnt 0x0
#endif
}

// Assemble a 16-element bf16 WMMA fragment from two contiguous 8-element
// (16-byte) runs. All callers pass p1 = p0 + 8 for A and B alike, so both
// hold one 16-long K-run, k = 16*hs + s (a consistent order, not the ISA table order).
__device__ __forceinline__ v16bf frag16(const __bf16* p0, const __bf16* p1) {
  v8bf lo = *(const v8bf*)p0;
  v8bf hi = *(const v8bf*)p1;
  v16bf r;
#pragma unroll
  for (int i = 0; i < 8; ++i) { r[i] = lo[i]; r[i + 8] = hi[i]; }
  return r;
}

__device__ __forceinline__ v8f wmma_bf16(v16bf a, v16bf b, v8f c) {
  // D(16x16,f32) = A(16x32,bf16) x B(32x16,bf16) + C
  return __builtin_amdgcn_wmma_f32_16x16x32_bf16(
      /*neg_a=*/false, a, /*neg_b=*/false, b,
      /*c_mod=*/(short)0, c, /*reuse_a=*/false, /*reuse_b=*/false);
}

// ---------------------------------------------------------------------------
// Weight transpose + fp32 -> bf16 convert:  dst[n][k] = (bf16) src[k][n]
// src: (K x N) fp32 row-major, dst: (N x K) bf16 row-major.
// block (32,8), grid (N/32, K/32)
// ---------------------------------------------------------------------------
__global__ void wcvt_t_kernel(const float* __restrict__ src,
                              __bf16* __restrict__ dst, int K, int N) {
  __shared__ float tile[32][33];
  const int tx = threadIdx.x, ty = threadIdx.y;
  const int n0 = blockIdx.x * 32, k0 = blockIdx.y * 32;
#pragma unroll
  for (int i = 0; i < 4; ++i)
    tile[ty * 4 + i][tx] = src[(size_t)(k0 + ty * 4 + i) * N + n0 + tx];
  __syncthreads();
#pragma unroll
  for (int i = 0; i < 4; ++i)
    dst[(size_t)(n0 + ty * 4 + i) * K + k0 + tx] = (__bf16)tile[tx][ty * 4 + i];
}

// ---------------------------------------------------------------------------
// LayerNorm over DIM=1024, fp32 in -> bf16 out.  grid = 4096, block = 256.
// ---------------------------------------------------------------------------
__global__ void ln_bf16_kernel(const float* __restrict__ x,
                               const float* __restrict__ gamma,
                               const float* __restrict__ beta,
                               __bf16* __restrict__ out) {
  const int row = blockIdx.x;
  const float* xr = x + (size_t)row * DIMX;
  const int t = threadIdx.x;
  float v[4], s = 0.f, s2 = 0.f;
#pragma unroll
  for (int i = 0; i < 4; ++i) {
    v[i] = xr[t + i * 256];
    s += v[i];
    s2 += v[i] * v[i];
  }
#pragma unroll
  for (int o = 16; o > 0; o >>= 1) {
    s  += __shfl_xor(s, o);
    s2 += __shfl_xor(s2, o);
  }
  __shared__ float ws[8], ws2[8];
  if ((t & 31) == 0) { ws[t >> 5] = s; ws2[t >> 5] = s2; }
  __syncthreads();
  s = 0.f; s2 = 0.f;
#pragma unroll
  for (int w = 0; w < 8; ++w) { s += ws[w]; s2 += ws2[w]; }
  const float mu  = s * (1.0f / DIMX);
  const float var = s2 * (1.0f / DIMX) - mu * mu;
  const float rs  = rsqrtf(var + 1e-5f);
  __bf16* orow = out + (size_t)row * DIMX;
#pragma unroll
  for (int i = 0; i < 4; ++i) {
    const int c = t + i * 256;
    orow[c] = (__bf16)((v[i] - mu) * rs * gamma[c] + beta[c]);
  }
}

// ---------------------------------------------------------------------------
// WMMA GEMM:  C(M,N) = A(M,K) @ Bt(N,K)^T  (A,Bt bf16 row-major)
// Block tile 128x256, 8 waves of 64x64 (4x4 WMMA accumulators).
// Double-buffered async LDS staging, one barrier per 64-K step.
// Epilogues: 0: ->bf16        1: +bias, exact GELU -> bf16
//            2: +resid ->f32  3: +bias, +resid -> f32
// grid = (N/256, M/128), block = 256.
// ---------------------------------------------------------------------------
__global__ __launch_bounds__(256) void gemm_wmma_kernel(
    const __bf16* __restrict__ A, const __bf16* __restrict__ Bt,
    void* __restrict__ Cout, const float* __restrict__ bias,
    const float* __restrict__ resid, int M, int N, int K, int epi) {
  __shared__ __bf16 As[2][128 * 72];   // 128 rows x 64 K, stride 72 (16B align)
  __shared__ __bf16 Bs[2][256 * 72];
  const int t    = threadIdx.x;
  const int lane = t & 31, wave = t >> 5;
  const int wm   = (wave >> 2) * 64, wn = (wave & 3) * 64;
  const int bm   = blockIdx.y * 128, bn = blockIdx.x * 256;
  const int m16  = lane & 15, hs = lane >> 4;

  v8f acc[4][4];
#pragma unroll
  for (int i = 0; i < 4; ++i)
#pragma unroll
    for (int j = 0; j < 4; ++j) acc[i][j] = v8f_zero();

  // issue one 128x64 A tile + 256x64 B tile into LDS buffer p
  auto stage = [&](int p, int kb) {
#pragma unroll
    for (int i = 0; i < 4; ++i) {
      const int c = t + i * 256;
      const int row = c >> 3, c8 = (c & 7) * 8;
      cp16_g2l(&A[(size_t)(bm + row) * K + kb + c8], &As[p][row * 72 + c8]);
    }
#pragma unroll
    for (int i = 0; i < 8; ++i) {
      const int c = t + i * 256;
      const int row = c >> 3, c8 = (c & 7) * 8;
      cp16_g2l(&Bt[(size_t)(bn + row) * K + kb + c8], &Bs[p][row * 72 + c8]);
    }
  };

  stage(0, 0);
  int p = 0;
  for (int kb = 0; kb < K; kb += 64, p ^= 1) {
    cp_fence();        // this wave's async copies done
    __syncthreads();   // everyone's copies published; prev buffer fully read
    if (kb + 64 < K) stage(p ^ 1, kb + 64);   // overlap next tile with compute
#pragma unroll
    for (int kk = 0; kk < 64; kk += 32) {
      v16bf af[4], bfr[4];
#pragma unroll
      for (int i = 0; i < 4; ++i) {
        const __bf16* pa = &As[p][(wm + i * 16 + m16) * 72 + kk + hs * 16];
        af[i] = frag16(pa, pa + 8);
      }
#pragma unroll
      for (int j = 0; j < 4; ++j) {
        const __bf16* pb = &Bs[p][(wn + j * 16 + m16) * 72 + kk + hs * 16];
        bfr[j] = frag16(pb, pb + 8);
      }
#pragma unroll
      for (int i = 0; i < 4; ++i)
#pragma unroll
        for (int j = 0; j < 4; ++j)
          acc[i][j] = wmma_bf16(af[i], bfr[j], acc[i][j]);
    }
  }

  // ---- epilogue (C/D layout: M = r + 8*hs, N = lane&15) ----
#pragma unroll
  for (int i = 0; i < 4; ++i) {
#pragma unroll
    for (int j = 0; j < 4; ++j) {
#pragma unroll
      for (int r = 0; r < 8; ++r) {
        const int row = bm + wm + i * 16 + r + hs * 8;
        const int col = bn + wn + j * 16 + m16;
        float v = acc[i][j][r];
        if (epi == 1) {
          v += bias[col];
          v = 0.5f * v * (1.0f + erff(v * 0.70710678118654752f));
        } else if (epi == 3) {
          v += bias[col];
        }
        if (epi == 2 || epi == 3) {
          v += resid[(size_t)row * N + col];
          ((float*)Cout)[(size_t)row * N + col] = v;
        } else {
          ((__bf16*)Cout)[(size_t)row * N + col] = (__bf16)v;
        }
      }
    }
  }
}

// ---------------------------------------------------------------------------
// Per-head V transpose: Vh (1024 x 64) -> Vt (64 x 1024), bf16.
// grid (32, 64): blockIdx.y = b*16+n, blockIdx.x = 32-row j tile.
// ---------------------------------------------------------------------------
__global__ void vhead_t_kernel(const __bf16* __restrict__ v,
                               __bf16* __restrict__ vt) {
  const int bn = blockIdx.y;
  const int jt = blockIdx.x * 32;
  const __bf16* src = v + (size_t)bn * (SEQX * HEADD);
  __bf16* dst = vt + (size_t)bn * (SEQX * HEADD);
  __shared__ __bf16 tile[32][72];
  const int t = threadIdx.x;
#pragma unroll
  for (int i = 0; i < 8; ++i) {
    const int e = t + i * 256;
    const int j = e >> 6, d = e & 63;
    tile[j][d] = src[(size_t)(jt + j) * HEADD + d];
  }
  __syncthreads();
#pragma unroll
  for (int i = 0; i < 8; ++i) {
    const int e = t + i * 256;
    const int d = e >> 5, j = e & 31;
    dst[(size_t)d * SEQX + jt + j] = tile[j][d];
  }
}

// ---------------------------------------------------------------------------
// Attention: per (b, head, 32-query tile).
// Phase 1: S = (Q Kh^T) * 1/8 via WMMA (wave w owns keys [128w,128w+128))
// Phase 2: row softmax in LDS fp32, P -> bf16 LDS
// Phase 3: O = P @ V via WMMA with per-wave key split + LDS reduction
// grid (32, 64), block 256.  LDS ~= 257 KB (<= 320 KB WGP budget).
// ---------------------------------------------------------------------------
#define QT 32
__global__ __launch_bounds__(256) void attn_wmma_kernel(
    const __bf16* __restrict__ q, const __bf16* __restrict__ k,
    const __bf16* __restrict__ vt, __bf16* __restrict__ attn) {
  __shared__ float  S[QT][1028];
  __shared__ __bf16 P[QT][1032];
  __shared__ float  Op[8][QT][64];

  const int bn = blockIdx.y;          // b*16 + n
  const int b = bn >> 4, n = bn & 15;
  const int qbase = blockIdx.x * QT;
  const __bf16* Qh = q  + (size_t)bn * (SEQX * HEADD);  // (1024 x 64)
  const __bf16* Kh = k  + (size_t)bn * (SEQX * HEADD);  // (1024 x 64)
  const __bf16* Vt = vt + (size_t)bn * (SEQX * HEADD);  // (64 x 1024)
  const int t = threadIdx.x, lane = t & 31, wave = t >> 5;
  const int m16 = lane & 15, hs = lane >> 4;
  const int kw = wave * 128;

  // ---- phase 1: scores ----
#pragma unroll
  for (int qi = 0; qi < 2; ++qi) {
    const __bf16* pq0 = &Qh[(size_t)(qbase + qi * 16 + m16) * HEADD + 0  + hs * 16];
    const __bf16* pq1 = &Qh[(size_t)(qbase + qi * 16 + m16) * HEADD + 32 + hs * 16];
    const v16bf aq0 = frag16(pq0, pq0 + 8);
    const v16bf aq1 = frag16(pq1, pq1 + 8);
#pragma unroll
    for (int kt = 0; kt < 8; ++kt) {
      const int kb = kw + kt * 16;
      const __bf16* pb0 = &Kh[(size_t)(kb + m16) * HEADD + 0  + hs * 16];
      const __bf16* pb1 = &Kh[(size_t)(kb + m16) * HEADD + 32 + hs * 16];
      v8f s = v8f_zero();
      s = wmma_bf16(aq0, frag16(pb0, pb0 + 8), s);
      s = wmma_bf16(aq1, frag16(pb1, pb1 + 8), s);
#pragma unroll
      for (int r = 0; r < 8; ++r)
        S[qi * 16 + r + hs * 8][kb + m16] = s[r] * 0.125f;
    }
  }
  __syncthreads();

  // ---- phase 2: softmax over 1024 keys (8 threads per row) ----
  {
    const int r = t >> 3, g = t & 7;
    float mx = -3.0e38f;
    for (int j = g * 128; j < g * 128 + 128; ++j) mx = fmaxf(mx, S[r][j]);
#pragma unroll
    for (int o = 4; o > 0; o >>= 1) mx = fmaxf(mx, __shfl_xor(mx, o));
    float sum = 0.f;
    for (int j = g * 128; j < g * 128 + 128; ++j) {
      const float e = __expf(S[r][j] - mx);
      S[r][j] = e;
      sum += e;
    }
#pragma unroll
    for (int o = 4; o > 0; o >>= 1) sum += __shfl_xor(sum, o);
    const float inv = 1.0f / sum;
    for (int j = g * 128; j < g * 128 + 128; ++j)
      P[r][j] = (__bf16)(S[r][j] * inv);
  }
  __syncthreads();

  // ---- phase 3: O = P @ V, each wave accumulates its 128-key slice ----
  v8f oacc[2][4];
#pragma unroll
  for (int qi = 0; qi < 2; ++qi)
#pragma unroll
    for (int nt = 0; nt < 4; ++nt) oacc[qi][nt] = v8f_zero();
#pragma unroll
  for (int ks = 0; ks < 4; ++ks) {
    const int kb = kw + ks * 32;
#pragma unroll
    for (int qi = 0; qi < 2; ++qi) {
      const __bf16* pa = &P[qi * 16 + m16][kb + hs * 16];
      const v16bf a = frag16(pa, pa + 8);
#pragma unroll
      for (int nt = 0; nt < 4; ++nt) {
        const __bf16* pb = &Vt[(size_t)(nt * 16 + m16) * SEQX + kb + hs * 16];
        oacc[qi][nt] = wmma_bf16(a, frag16(pb, pb + 8), oacc[qi][nt]);
      }
    }
  }
#pragma unroll
  for (int qi = 0; qi < 2; ++qi)
#pragma unroll
    for (int nt = 0; nt < 4; ++nt)
#pragma unroll
      for (int r = 0; r < 8; ++r)
        Op[wave][qi * 16 + r + hs * 8][nt * 16 + m16] = oacc[qi][nt][r];
  __syncthreads();

  // ---- reduce 8 wave partials, write to (b, i, n*64+d) ----
#pragma unroll
  for (int i = 0; i < 8; ++i) {
    const int e = t + i * 256;          // 0..2047
    const int qr = e >> 6, d = e & 63;
    float s = 0.f;
#pragma unroll
    for (int w = 0; w < 8; ++w) s += Op[w][qr][d];
    attn[(size_t)(b * SEQX + qbase + qr) * DIMX + n * HEADD + d] = (__bf16)s;
  }
}

// ---------------------------------------------------------------------------
// Launch
// ---------------------------------------------------------------------------
extern "C" void kernel_launch(void* const* d_in, const int* in_sizes, int n_in,
                              void* d_out, int out_size, void* d_ws,
                              size_t ws_size, hipStream_t stream) {
  (void)in_sizes; (void)n_in; (void)out_size; (void)ws_size;
  const float* x   = (const float*)d_in[0];
  const float* Wq  = (const float*)d_in[1];
  const float* Wk  = (const float*)d_in[2];
  const float* Wv  = (const float*)d_in[3];
  const float* Wo  = (const float*)d_in[4];
  const float* W1  = (const float*)d_in[5];
  const float* b1  = (const float*)d_in[6];
  const float* W2  = (const float*)d_in[7];
  const float* b2  = (const float*)d_in[8];
  const float* g1  = (const float*)d_in[9];
  const float* bb1 = (const float*)d_in[10];
  const float* g2  = (const float*)d_in[11];
  const float* bb2 = (const float*)d_in[12];
  float* out = (float*)d_out;

  char* base = (char*)d_ws;
  size_t off = 0;
  auto take = [&](size_t bytes) -> char* {
    char* p = base + off;
    off += (bytes + 255) & ~(size_t)255;
    return p;
  };
  const size_t NB = (size_t)ROWSX * DIMX;           // 4M elements
  __bf16* WqT = (__bf16*)take(DIMX * DIMX * 2);
  __bf16* WkT = (__bf16*)take(DIMX * DIMX * 2);
  __bf16* WvT = (__bf16*)take(DIMX * DIMX * 2);
  __bf16* WoT = (__bf16*)take(DIMX * DIMX * 2);
  __bf16* W1T = (__bf16*)take((size_t)MLPX * DIMX * 2);
  __bf16* W2T = (__bf16*)take((size_t)DIMX * MLPX * 2);
  __bf16* h1  = (__bf16*)take(NB * 2);              // reused as h2
  __bf16* qb  = (__bf16*)take(NB * 2);              // q/k/v/vt region reused as f1
  __bf16* kb  = (__bf16*)take(NB * 2);
  __bf16* vb  = (__bf16*)take(NB * 2);
  __bf16* vtb = (__bf16*)take(NB * 2);
  __bf16* ab  = (__bf16*)take(NB * 2);              // attention out, bf16
  float*  x2  = (float*)take(NB * 4);
  __bf16* h2  = h1;
  __bf16* f1  = qb;                                 // (4096 x 4096) bf16

  const dim3 bt(32, 8);
  // weights -> bf16, transposed (output-major)
  wcvt_t_kernel<<<dim3(32, 32), bt, 0, stream>>>(Wq, WqT, DIMX, DIMX);
  wcvt_t_kernel<<<dim3(32, 32), bt, 0, stream>>>(Wk, WkT, DIMX, DIMX);
  wcvt_t_kernel<<<dim3(32, 32), bt, 0, stream>>>(Wv, WvT, DIMX, DIMX);
  wcvt_t_kernel<<<dim3(32, 32), bt, 0, stream>>>(Wo, WoT, DIMX, DIMX);
  wcvt_t_kernel<<<dim3(128, 32), bt, 0, stream>>>(W1, W1T, DIMX, MLPX);
  wcvt_t_kernel<<<dim3(32, 128), bt, 0, stream>>>(W2, W2T, MLPX, DIMX);

  // attention sublayer
  ln_bf16_kernel<<<ROWSX, 256, 0, stream>>>(x, g1, bb1, h1);
  gemm_wmma_kernel<<<dim3(4, 32), 256, 0, stream>>>(h1, WqT, qb, nullptr, nullptr, ROWSX, DIMX, DIMX, 0);
  gemm_wmma_kernel<<<dim3(4, 32), 256, 0, stream>>>(h1, WkT, kb, nullptr, nullptr, ROWSX, DIMX, DIMX, 0);
  gemm_wmma_kernel<<<dim3(4, 32), 256, 0, stream>>>(h1, WvT, vb, nullptr, nullptr, ROWSX, DIMX, DIMX, 0);
  vhead_t_kernel<<<dim3(32, 64), 256, 0, stream>>>(vb, vtb);
  attn_wmma_kernel<<<dim3(SEQX / QT, BATCHX * NHEADS), 256, 0, stream>>>(qb, kb, vtb, ab);
  gemm_wmma_kernel<<<dim3(4, 32), 256, 0, stream>>>(ab, WoT, x2, nullptr, x, ROWSX, DIMX, DIMX, 2);

  // FFN sublayer
  ln_bf16_kernel<<<ROWSX, 256, 0, stream>>>(x2, g2, bb2, h2);
  gemm_wmma_kernel<<<dim3(16, 32), 256, 0, stream>>>(h2, W1T, f1, b1, nullptr, ROWSX, MLPX, DIMX, 1);
  gemm_wmma_kernel<<<dim3(4, 32), 256, 0, stream>>>(f1, W2T, out, b2, x2, ROWSX, DIMX, MLPX, 3);
}
